// MemoryCore_50251117363637
// MI455X (gfx1250) — matrix operand order repaired and checked by reading, not compiled, not run on hardware
//
#include <hip/hip_runtime.h>

typedef unsigned short u16;
typedef unsigned int   u32;
typedef unsigned long long u64;
typedef __attribute__((ext_vector_type(16))) __bf16 v16bf;
typedef __attribute__((ext_vector_type(8)))  float  v8f;

// TDM descriptor groups (6-arg builtin form: g0 u32x4, g1 i32x8, g2 i32x4, g3 i32x4, i32x8, cpol)
typedef __attribute__((ext_vector_type(4))) unsigned int tdm_g0_t;
typedef __attribute__((ext_vector_type(8))) int          tdm_g1_t;
typedef __attribute__((ext_vector_type(4))) int          tdm_g2_t;

#define DDIM    512
#define LS      520          // padded LDS row stride in elements (1040B -> 4-bank rotate/row)
#define BN      128          // embedding rows per workgroup
#define BM      64           // bank rows per LDS chunk
#define MSPLIT  8            // M split across grid.x (combined via atomicMin)
#define NROWS   8192
#define MROWS   65536
#define MCHUNK  (MROWS / MSPLIT)   // 8192
#define NCHUNKS (MCHUNK / BM)      // 128

__device__ __forceinline__ u16 f2bf(float f) {
    u32 u = __float_as_uint(f);
    u32 r = (u + 0x7FFFu + ((u >> 16) & 1u)) >> 16;   // round-to-nearest-even
    return (u16)r;
}

union FragBF { uint4 u[2]; v16bf v; };

// ---------------------------------------------------------------------------
// TDM: async-load a (tileRows x 512) bf16 tile, row-major stride 512, into LDS
// with hardware row padding of 4 DWORDs per 256 DWORDs -> LDS stride 520 elems.
// Issued by a single wave; tracked via TENSORcnt.
// ---------------------------------------------------------------------------
__device__ __forceinline__ void tdm_load_tile(u32 ldsOff, const u16* gptr, int tileRows) {
    const u64 ga = (u64)(size_t)gptr;
    tdm_g0_t g0;
    g0[0] = 1u;                                             // count=1, no gather
    g0[1] = ldsOff;                                         // lds_addr (bytes)
    g0[2] = (u32)ga;                                        // global_addr[31:0]
    g0[3] = ((u32)(ga >> 32) & 0x01FFFFFFu) | (2u << 30);   // global_addr[56:32] | type=2
    tdm_g1_t g1;
    g1[0] = (int)((1u << 16)      // data_size = 2 bytes
                | (1u << 20)      // pad_enable
                | (7u << 22)      // pad_interval: 256 DWORDs (= one 512-elem row)
                | (3u << 25));    // pad_amount: 4 DWORDs (= 8 elems -> stride 520)
    g1[1] = (int)(512u << 16);    // tensor_dim0 = 512 (low 16 in bits 63:48)
    g1[2] = 0;                    // tensor_dim0 hi / tensor_dim1 low16 (65536 -> 0)
    g1[3] = (int)(1u | (512u << 16)); // tensor_dim1 hi16 = 1 (65536); tile_dim0 = 512
    g1[4] = tileRows;             // tile_dim1 = rows; tile_dim2 = 0
    g1[5] = 512;                  // tensor_dim0_stride[31:0] = 512 elems
    g1[6] = 0;                    // strides hi
    g1[7] = 0;
    tdm_g2_t gz;
    gz[0] = 0; gz[1] = 0; gz[2] = 0; gz[3] = 0;             // 2-D tile: groups 2/3 unused
    tdm_g1_t gz8;
    gz8[0] = 0; gz8[1] = 0; gz8[2] = 0; gz8[3] = 0;
    gz8[4] = 0; gz8[5] = 0; gz8[6] = 0; gz8[7] = 0;
    __builtin_amdgcn_tensor_load_to_lds(g0, g1, gz, gz, gz8, 0);
}

// ---------------------------------------------------------------------------
// Kernel 1/2: fp32 -> bf16 conversion + exact fp32 row squared-norms.
// One wave per row (8 rows per 256-thread block).
// ---------------------------------------------------------------------------
__global__ __launch_bounds__(256) void k_conv(const float* __restrict__ src,
                                              u16* __restrict__ dst,
                                              float* __restrict__ sq,
                                              int rows) {
    const int wave = threadIdx.x >> 5;
    const int lane = threadIdx.x & 31;
    const int row  = blockIdx.x * 8 + wave;
    if (row >= rows) return;

    const float* s = src + (size_t)row * DDIM;
    u16*         d = dst + (size_t)row * DDIM;

    float acc = 0.f;
#pragma unroll
    for (int j = 0; j < 4; ++j) {
        const int idx = j * 128 + lane * 4;
        float4 v = *reinterpret_cast<const float4*>(s + idx);
        acc += v.x * v.x + v.y * v.y + v.z * v.z + v.w * v.w;
        union { u16 h[4]; uint2 u; } pk;
        pk.h[0] = f2bf(v.x); pk.h[1] = f2bf(v.y);
        pk.h[2] = f2bf(v.z); pk.h[3] = f2bf(v.w);
        *reinterpret_cast<uint2*>(d + idx) = pk.u;
    }
#pragma unroll
    for (int off = 16; off > 0; off >>= 1) acc += __shfl_xor(acc, off, 32);
    if (lane == 0) sq[row] = acc;
}

// ---------------------------------------------------------------------------
// Kernel 3: init running min(dist^2) to +inf (bit pattern works as signed-int min)
// ---------------------------------------------------------------------------
__global__ void k_init(u32* __restrict__ minsq, int n) {
    int i = blockIdx.x * blockDim.x + threadIdx.x;
    if (i < n) minsq[i] = 0x7F800000u;
}

// ---------------------------------------------------------------------------
// Kernel 4: main tiled bf16 WMMA distance kernel with TDM-fed LDS.
// grid = (MSPLIT, NROWS/BN). Block = 256 threads = 8 waves.
// Wave w owns rows [w*16, w*16+16) of the block's 128-row tile, and
// computes 4 16x16 tiles (64 bank columns) per LDS chunk.
// ---------------------------------------------------------------------------
__global__ __launch_bounds__(256, 1) void k_main(const u16* __restrict__ embB,
                                                 const u16* __restrict__ bankB,
                                                 const float* __restrict__ xsq,
                                                 const float* __restrict__ msq,
                                                 u32* __restrict__ minsq) {
    __shared__ u16 sE[BN * LS];
    __shared__ u16 sB[2][BM * LS];

    const int tid  = threadIdx.x;
    const int wave = tid >> 5;
    const int lane = tid & 31;
    const int l16  = lane & 15;
    const int hi   = lane >> 4;     // 0: lanes 0-15, 1: lanes 16-31

    const int rb = blockIdx.y * BN;       // embedding row base
    const int mb = blockIdx.x * MCHUNK;   // memory-bank row base for this chunk

    // ---- kick off async TDM loads: embedding tile + first bank chunk ----
    if (wave == 0) {
        tdm_load_tile((u32)(size_t)&sE[0],    embB  + (size_t)rb * DDIM, BN);
        tdm_load_tile((u32)(size_t)&sB[0][0], bankB + (size_t)mb * DDIM, BM);
    }

    // per-lane exact ||x||^2 for the 8 rows this lane accumulates (C layout: M = r + 8*hi)
    float xr[8];
#pragma unroll
    for (int r = 0; r < 8; ++r) xr[r] = xsq[rb + wave * 16 + r + 8 * hi];

    float run[8];
#pragma unroll
    for (int r = 0; r < 8; ++r) run[r] = 3.4e38f;

    // A fragment base: lane holds row (wave*16 + l16); K-halves interleaved per ISA layout
    const u16* eBase = &sE[(wave * 16 + l16) * LS + (hi ? 8 : 0)];

    for (int c = 0; c < NCHUNKS; ++c) {
        // TDM ops are in-order per wave: wave 0 drains its tensor queue, then the
        // barrier publishes the freshly written LDS tile to all 8 waves.
        __builtin_amdgcn_s_wait_tensorcnt((short)0);
        __syncthreads();   // also guarantees compute on buf[(c+1)&1] (iter c-1) is done
        if (wave == 0 && c + 1 < NCHUNKS)
            tdm_load_tile((u32)(size_t)&sB[(c + 1) & 1][0],
                          bankB + (size_t)(mb + (c + 1) * BM) * DDIM, BM);

        const u16* bb = sB[c & 1];

        v8f acc[4];
#pragma unroll
        for (int t = 0; t < 4; ++t)
#pragma unroll
            for (int e = 0; e < 8; ++e) acc[t][e] = 0.f;

#pragma unroll 4
        for (int k = 0; k < DDIM; k += 32) {
            FragBF a;
            a.u[0] = *reinterpret_cast<const uint4*>(eBase + k);        // K +0..7  / +8..15
            a.u[1] = *reinterpret_cast<const uint4*>(eBase + k + 16);   // K +16..23 / +24..31
#pragma unroll
            for (int t = 0; t < 4; ++t) {
                FragBF b;   // B layout: lane holds col l16, K-halves interleaved per ISA layout (same as A)
                const u16* p = bb + (t * 16 + l16) * LS + (hi ? 8 : 0) + k;
                b.u[0] = *reinterpret_cast<const uint4*>(p);
                b.u[1] = *reinterpret_cast<const uint4*>(p + 16);
                acc[t] = __builtin_amdgcn_wmma_f32_16x16x32_bf16(
                    false, a.v, false, b.v, (short)0, acc[t], false, false);
            }
        }

        // fold this 64-column chunk into the running per-row minima
        const int mcb = mb + c * BM;
#pragma unroll
        for (int t = 0; t < 4; ++t) {
            const float ms = msq[mcb + t * 16 + l16];
#pragma unroll
            for (int r = 0; r < 8; ++r) {
                float d2 = fmaxf(xr[r] + ms - 2.0f * acc[t][r], 0.0f);
                run[r] = fminf(run[r], d2);
            }
        }
    }

    // min across the 16 lanes of each half (xor stays inside each 16-lane group)
#pragma unroll
    for (int r = 0; r < 8; ++r) {
        float v = run[r];
#pragma unroll
        for (int off = 1; off < 16; off <<= 1) v = fminf(v, __shfl_xor(v, off, 32));
        run[r] = v;
    }
    if (l16 == 0) {
#pragma unroll
        for (int r = 0; r < 8; ++r) {
            const int row = rb + wave * 16 + r + 8 * hi;
            atomicMin(reinterpret_cast<int*>(&minsq[row]), __float_as_int(run[r]));
        }
    }
}

// ---------------------------------------------------------------------------
// Kernel 5: finalize -> sqrt of the per-row min dist^2
// ---------------------------------------------------------------------------
__global__ void k_fin(const u32* __restrict__ minsq, float* __restrict__ out, int n) {
    int i = blockIdx.x * blockDim.x + threadIdx.x;
    if (i < n) out[i] = sqrtf(fmaxf(__uint_as_float(minsq[i]), 0.0f));
}

// ---------------------------------------------------------------------------
extern "C" void kernel_launch(void* const* d_in, const int* in_sizes, int n_in,
                              void* d_out, int out_size, void* d_ws, size_t ws_size,
                              hipStream_t stream) {
    (void)in_sizes; (void)n_in; (void)out_size; (void)ws_size;

    const float* emb  = (const float*)d_in[0];   // [8192, 512]
    const float* bank = (const float*)d_in[1];   // [65536, 512]
    float* out = (float*)d_out;                  // [8192, 1]

    char* ws = (char*)d_ws;
    u16*   bankB = (u16*)(ws);                                              // 64 MB
    u16*   embB  = (u16*)(ws + (size_t)MROWS * DDIM * sizeof(u16));         // +8 MB
    float* msqp  = (float*)(ws + (size_t)(MROWS + NROWS) * DDIM * sizeof(u16));
    float* xsqp  = (float*)((char*)msqp + (size_t)MROWS * sizeof(float));
    u32*   minp  = (u32*)((char*)xsqp + (size_t)NROWS * sizeof(float));

    k_conv<<<MROWS / 8, 256, 0, stream>>>(bank, bankB, msqp, MROWS);
    k_conv<<<NROWS / 8, 256, 0, stream>>>(emb,  embB,  xsqp, NROWS);
    k_init<<<NROWS / 256, 256, 0, stream>>>(minp, NROWS);
    k_main<<<dim3(MSPLIT, NROWS / BN, 1), 256, 0, stream>>>(embB, bankB, xsqp, msqp, minp);
    k_fin<<<NROWS / 256, 256, 0, stream>>>(minp, out, NROWS);
}
